// HRNN_51118700757067
// MI455X (gfx1250) — compile-verified, matrix operand order checked by reading, not run on hardware
//
#include <hip/hip_runtime.h>

// ---------------------------------------------------------------------------
// Problem constants (from reference)
// ---------------------------------------------------------------------------
#define E_    16
#define L_    3
#define D_IN  32
#define D_    512
#define H_FF  2048
#define D_ENC 512
#define H_DEC 2048
#define D_OUT 1024
#define T_    128
#define CAT_  (L_ * D_)      // 1536
#define D_CAT (E_ * D_ENC)   // 8192

typedef __attribute__((ext_vector_type(16))) __bf16 v16bf;
typedef __attribute__((ext_vector_type(8)))  __bf16 v8bf;
typedef __attribute__((ext_vector_type(8)))  float  v8f;

// ---------------------------------------------------------------------------
// Weight conversion: fp32 [B][K][N] -> bf16 transposed [B][N][K]
// (one-shot per launch; makes B-fragment loads contiguous b128 in K)
// ---------------------------------------------------------------------------
__global__ void cvt_transpose_bf16(const float* __restrict__ src,
                                   __bf16* __restrict__ dst,
                                   int Kd, int Nd, long long total) {
    long long stride = (long long)gridDim.x * blockDim.x;
    long long kn = (long long)Kd * Nd;
    for (long long idx = (long long)blockIdx.x * blockDim.x + threadIdx.x;
         idx < total; idx += stride) {
        long long mat = idx / kn;
        long long r   = idx - mat * kn;
        int kk = (int)(r / Nd);
        int nn = (int)(r - (long long)kk * Nd);
        dst[mat * kn + (long long)nn * Kd + kk] = (__bf16)src[idx];
    }
}

// ---------------------------------------------------------------------------
// Phase 1: the sequential recurrence. 1 block per encoder, 512 threads
// (one per state element). Hidden state double-buffered in LDS (fp32).
// Emits the concatenated states [E][T][L*D] in bf16 for the phase-2 GEMMs.
// ---------------------------------------------------------------------------
__global__ __launch_bounds__(512)
void rnn_scan_kernel(const float* __restrict__ x,
                     const float* __restrict__ W_in0,
                     const float* __restrict__ Wh0,
                     const float* __restrict__ b0,
                     const float* __restrict__ W_in_rest,
                     const float* __restrict__ Wh_rest,
                     const float* __restrict__ b_rest,
                     __bf16* __restrict__ cat_out) {
    const int e = blockIdx.x;
    const int k = threadIdx.x;            // 0..511

    __shared__ float h[2][L_][D_];        // double-buffered hidden state
    __shared__ float xt[D_IN];

    for (int l = 0; l < L_; ++l) h[0][l][k] = 0.0f;

    const float* Wi0e = W_in0 + (size_t)e * D_IN * D_;
    const float* Wh0e = Wh0   + (size_t)e * D_ * D_;
    const float  b0k  = b0[(size_t)e * D_ + k];
    const float* Wir[L_ - 1];
    const float* Whr[L_ - 1];
    float        brk[L_ - 1];
    for (int l = 0; l < L_ - 1; ++l) {
        Wir[l] = W_in_rest + (size_t)(e * (L_ - 1) + l) * D_ * D_;
        Whr[l] = Wh_rest   + (size_t)(e * (L_ - 1) + l) * D_ * D_;
        brk[l] = b_rest[(size_t)(e * (L_ - 1) + l) * D_ + k];
    }
    __syncthreads();

    for (int t = 0; t < T_; ++t) {
        const int cur = t & 1, prv = cur ^ 1;
        if (k < D_IN) xt[k] = x[t * D_IN + k];
        __syncthreads();

        // ---- layer 0: tanh(x_t @ W_in0 + h_prev0 @ Wh0 + b0)
        float acc = b0k;
        #pragma unroll 8
        for (int i = 0; i < D_IN; ++i) acc += xt[i] * Wi0e[i * D_ + k];
        #pragma unroll 8
        for (int d = 0; d < D_; ++d)   acc += h[prv][0][d] * Wh0e[d * D_ + k];
        float h0 = tanhf(acc);
        h[cur][0][k] = h0;
        cat_out[((size_t)e * T_ + t) * CAT_ + k] = (__bf16)h0;
        __syncthreads();

        // ---- layers 1..L-1: tanh(h_new[l-1] @ W_in + h_prev[l] @ Wh + b)
        for (int l = 1; l < L_; ++l) {
            float a2 = brk[l - 1];
            const float* Wi = Wir[l - 1];
            const float* Wr = Whr[l - 1];
            #pragma unroll 8
            for (int d = 0; d < D_; ++d) a2 += h[cur][l - 1][d] * Wi[d * D_ + k];
            #pragma unroll 8
            for (int d = 0; d < D_; ++d) a2 += h[prv][l][d] * Wr[d * D_ + k];
            float hl = tanhf(a2);
            h[cur][l][k] = hl;
            cat_out[((size_t)e * T_ + t) * CAT_ + l * D_ + k] = (__bf16)hl;
            __syncthreads();
        }
    }
}

// ---------------------------------------------------------------------------
// Phase 2: bf16 WMMA GEMM, C[M,N] = act(A[M,K] @ B[K,N] + bias).
// B supplied pre-transposed as Bt[N][K].  Register-tiled: each wave owns a
// 32x32 output tile (2 A frags + 2 B frags -> 4 WMMAs per K-step), 8 waves
// (256 threads) per block -> 256 N-columns per block.  Encoder batch in
// gridDim.z with per-batch strides.
//   grid.x = N/256, grid.y = M/32, grid.z = batch
// Fragment layouts per CDNA5 ISA 7.12.2 (16-bit A 16x32; 32-bit C/D 16x16).
// ---------------------------------------------------------------------------
#define ACT_NONE 0
#define ACT_TANH 1
#define ACT_GELU 2

__global__ __launch_bounds__(256)
void wmma_gemm_bf16(const __bf16* __restrict__ A, int lda, long long strideA,
                    const __bf16* __restrict__ Bt, int K, long long strideB,
                    const float* __restrict__ bias, long long strideBias,
                    float* __restrict__ outF,           // fp32 out (or null)
                    __bf16* __restrict__ outB,          // bf16 out (or null)
                    int ldo, int out_off, int out_off_stride, long long strideOut,
                    int act) {
    const int lane = threadIdx.x & 31;
    const int wave = threadIdx.x >> 5;
    const int r    = lane & 15;           // row-in-tile (A) / col-in-tile (B)
    const int half = lane >> 4;           // K-half selector
    const int bz   = blockIdx.z;
    const int n0   = (blockIdx.x * 8 + wave) * 32;
    const int m0   = blockIdx.y * 32;

    const __bf16* Ae = A  + (long long)bz * strideA;
    const __bf16* Be = Bt + (long long)bz * strideB;

    const __bf16* a0row = Ae + (size_t)(m0 + r)      * lda + half * 8;
    const __bf16* a1row = Ae + (size_t)(m0 + 16 + r) * lda + half * 8;
    const __bf16* b0row = Be + (size_t)(n0 + r)      * K   + half * 8;
    const __bf16* b1row = Be + (size_t)(n0 + 16 + r) * K   + half * 8;

    v8f c00 = {}, c01 = {}, c10 = {}, c11 = {};
    for (int k0 = 0; k0 < K; k0 += 32) {
        // prefetch ahead in the K streams (global_prefetch_b8)
        __builtin_prefetch(a0row + k0 + 256, 0, 0);
        __builtin_prefetch(a1row + k0 + 256, 0, 0);
        __builtin_prefetch(b0row + k0 + 256, 0, 0);
        __builtin_prefetch(b1row + k0 + 256, 0, 0);

        v8bf a0l = *(const v8bf*)(a0row + k0);
        v8bf a0h = *(const v8bf*)(a0row + k0 + 16);
        v8bf a1l = *(const v8bf*)(a1row + k0);
        v8bf a1h = *(const v8bf*)(a1row + k0 + 16);
        v8bf b0l = *(const v8bf*)(b0row + k0);
        v8bf b0h = *(const v8bf*)(b0row + k0 + 16);
        v8bf b1l = *(const v8bf*)(b1row + k0);
        v8bf b1h = *(const v8bf*)(b1row + k0 + 16);
        v16bf A0 = __builtin_shufflevector(a0l, a0h, 0,1,2,3,4,5,6,7,
                                                     8,9,10,11,12,13,14,15);
        v16bf A1 = __builtin_shufflevector(a1l, a1h, 0,1,2,3,4,5,6,7,
                                                     8,9,10,11,12,13,14,15);
        v16bf B0 = __builtin_shufflevector(b0l, b0h, 0,1,2,3,4,5,6,7,
                                                     8,9,10,11,12,13,14,15);
        v16bf B1 = __builtin_shufflevector(b1l, b1h, 0,1,2,3,4,5,6,7,
                                                     8,9,10,11,12,13,14,15);
        // (neg_a, A, neg_b, B, c_mod, C, reuse_a, reuse_b)
        c00 = __builtin_amdgcn_wmma_f32_16x16x32_bf16(false, A0, false, B0,
                                                      (short)0, c00, false, false);
        c01 = __builtin_amdgcn_wmma_f32_16x16x32_bf16(false, A0, false, B1,
                                                      (short)0, c01, false, false);
        c10 = __builtin_amdgcn_wmma_f32_16x16x32_bf16(false, A1, false, B0,
                                                      (short)0, c10, false, false);
        c11 = __builtin_amdgcn_wmma_f32_16x16x32_bf16(false, A1, false, B1,
                                                      (short)0, c11, false, false);
    }

    // Epilogue: C/D layout -> lane holds (M = mbase + v + 8*half, N = nbase + r)
    const float*  biasb = bias ? bias + (long long)bz * strideBias : nullptr;
    float*        outFb = outF ? outF + (long long)bz * strideOut : nullptr;
    __bf16*       outBb = outB ? outB + (long long)bz * strideOut : nullptr;
    const int     ocol  = out_off + out_off_stride * bz;

    v8f accs[4] = {c00, c01, c10, c11};
    #pragma unroll
    for (int mi = 0; mi < 2; ++mi) {
        #pragma unroll
        for (int ni = 0; ni < 2; ++ni) {
            v8f c = accs[mi * 2 + ni];
            const int n = n0 + ni * 16 + r;
            #pragma unroll
            for (int v = 0; v < 8; ++v) {
                const int m = m0 + mi * 16 + v + 8 * half;
                float val = c[v];
                if (biasb) val += biasb[n];
                if (act == ACT_TANH) {
                    val = tanhf(val);
                } else if (act == ACT_GELU) {
                    // jax.nn.gelu (tanh approximation)
                    float x3 = val * val * val;
                    val = 0.5f * val * (1.0f + tanhf(0.7978845608028654f *
                                                     (val + 0.044715f * x3)));
                }
                const size_t o = (size_t)m * ldo + ocol + n;
                if (outFb) outFb[o] = val;
                else       outBb[o] = (__bf16)val;
            }
        }
    }
}

// ---------------------------------------------------------------------------
// Host-side orchestration
// ---------------------------------------------------------------------------
extern "C" void kernel_launch(void* const* d_in, const int* in_sizes, int n_in,
                              void* d_out, int out_size, void* d_ws, size_t ws_size,
                              hipStream_t stream) {
    const float* x         = (const float*)d_in[0];
    const float* W_in0     = (const float*)d_in[1];
    const float* Wh0       = (const float*)d_in[2];
    const float* b0        = (const float*)d_in[3];
    const float* W_in_rest = (const float*)d_in[4];
    const float* Wh_rest   = (const float*)d_in[5];
    const float* b_rest    = (const float*)d_in[6];
    const float* W_ff1     = (const float*)d_in[7];
    const float* b_ff1     = (const float*)d_in[8];
    const float* W_ff2     = (const float*)d_in[9];
    const float* b_ff2     = (const float*)d_in[10];
    const float* W_d0      = (const float*)d_in[11];
    const float* b_d0      = (const float*)d_in[12];
    const float* W_dmid    = (const float*)d_in[13];
    const float* b_dmid    = (const float*)d_in[14];
    const float* W_dout    = (const float*)d_in[15];
    const float* b_dout    = (const float*)d_in[16];
    float* out = (float*)d_out;

    // ---- workspace carve-up (256B aligned) ----
    char*  ws  = (char*)d_ws;
    size_t off = 0;
    auto alloc = [&](size_t bytes) -> char* {
        char* p = ws + off;
        off = (off + bytes + 255) & ~(size_t)255;
        return p;
    };
    __bf16* Wff1t = (__bf16*)alloc((size_t)E_ * CAT_ * H_FF   * 2);
    __bf16* Wff2t = (__bf16*)alloc((size_t)E_ * H_FF * D_ENC  * 2);
    __bf16* Wd0t  = (__bf16*)alloc((size_t)D_CAT * H_DEC      * 2);
    __bf16* Wdmt  = (__bf16*)alloc((size_t)2 * H_DEC * H_DEC  * 2);
    __bf16* Wdot  = (__bf16*)alloc((size_t)H_DEC * D_OUT      * 2);
    __bf16* catb  = (__bf16*)alloc((size_t)E_ * T_ * CAT_     * 2);
    __bf16* ffact = (__bf16*)alloc((size_t)E_ * T_ * H_FF     * 2);
    __bf16* encc  = (__bf16*)alloc((size_t)T_ * D_CAT         * 2);
    __bf16* z0    = (__bf16*)alloc((size_t)T_ * H_DEC         * 2);
    __bf16* z1    = (__bf16*)alloc((size_t)T_ * H_DEC         * 2);
    __bf16* z2    = (__bf16*)alloc((size_t)T_ * H_DEC         * 2);

    // ---- weight conversion + transpose (bf16, [N][K]) ----
    const dim3 cb(256), cg(2048);
    cvt_transpose_bf16<<<cg, cb, 0, stream>>>(W_ff1,  Wff1t, CAT_,  H_FF,
                                              (long long)E_ * CAT_ * H_FF);
    cvt_transpose_bf16<<<cg, cb, 0, stream>>>(W_ff2,  Wff2t, H_FF,  D_ENC,
                                              (long long)E_ * H_FF * D_ENC);
    cvt_transpose_bf16<<<cg, cb, 0, stream>>>(W_d0,   Wd0t,  D_CAT, H_DEC,
                                              (long long)D_CAT * H_DEC);
    cvt_transpose_bf16<<<cg, cb, 0, stream>>>(W_dmid, Wdmt,  H_DEC, H_DEC,
                                              (long long)2 * H_DEC * H_DEC);
    cvt_transpose_bf16<<<cg, cb, 0, stream>>>(W_dout, Wdot,  H_DEC, D_OUT,
                                              (long long)H_DEC * D_OUT);

    // ---- phase 1: sequential recurrence (fp32, L2-resident weights) ----
    rnn_scan_kernel<<<E_, D_, 0, stream>>>(x, W_in0, Wh0, b0,
                                           W_in_rest, Wh_rest, b_rest, catb);

    // ---- phase 2: batched WMMA GEMMs over T=128 rows ----
    // FF1: [128,1536] @ [1536,2048] + b -> GELU, batched over encoders (z)
    wmma_gemm_bf16<<<dim3(H_FF / 256, T_ / 32, E_), 256, 0, stream>>>(
        catb, CAT_, (long long)T_ * CAT_,
        Wff1t, CAT_, (long long)CAT_ * H_FF,
        b_ff1, H_FF,
        nullptr, ffact, H_FF, 0, 0, (long long)T_ * H_FF, ACT_GELU);
    // FF2: [128,2048] @ [2048,512] + b -> scatter into concat [128, e*512 + n]
    wmma_gemm_bf16<<<dim3(D_ENC / 256, T_ / 32, E_), 256, 0, stream>>>(
        ffact, H_FF, (long long)T_ * H_FF,
        Wff2t, H_FF, (long long)H_FF * D_ENC,
        b_ff2, D_ENC,
        nullptr, encc, D_CAT, 0, D_ENC, 0, ACT_NONE);
    // decoder layer 0: tanh([128,8192] @ [8192,2048] + b)
    wmma_gemm_bf16<<<dim3(H_DEC / 256, T_ / 32, 1), 256, 0, stream>>>(
        encc, D_CAT, 0, Wd0t, D_CAT, 0, b_d0, 0,
        nullptr, z0, H_DEC, 0, 0, 0, ACT_TANH);
    // decoder mids: tanh([128,2048] @ [2048,2048] + b), x2 (sequential)
    wmma_gemm_bf16<<<dim3(H_DEC / 256, T_ / 32, 1), 256, 0, stream>>>(
        z0, H_DEC, 0, Wdmt, H_DEC, 0, b_dmid, 0,
        nullptr, z1, H_DEC, 0, 0, 0, ACT_TANH);
    wmma_gemm_bf16<<<dim3(H_DEC / 256, T_ / 32, 1), 256, 0, stream>>>(
        z1, H_DEC, 0, Wdmt + (size_t)H_DEC * H_DEC, H_DEC, 0, b_dmid + H_DEC, 0,
        nullptr, z2, H_DEC, 0, 0, 0, ACT_TANH);
    // decoder out: [128,2048] @ [2048,1024] + b -> fp32 d_out
    wmma_gemm_bf16<<<dim3(D_OUT / 256, T_ / 32, 1), 256, 0, stream>>>(
        z2, H_DEC, 0, Wdot, H_DEC, 0, b_dout, 0,
        out, nullptr, D_OUT, 0, 0, 0, ACT_NONE);
}
